// MultilayerGRU_50337016709609
// MI455X (gfx1250) — matrix operand order repaired and checked by reading, not compiled, not run on hardware
//
#include <hip/hip_runtime.h>

// ---------------------------------------------------------------------------
// MultilayerGRU for MI455X (gfx1250, wave32, WMMA bf16 16x16x32, fp32 accum)
// ---------------------------------------------------------------------------
// B=32, S=512, I=256, H=1024, O=256
//   - input/output projections: batched WMMA GEMMs, 2 N-tiles per wave
//     (A-fragment reuse, dual accumulator chains)
//   - recurrence: 2 kernels per time step (z/r, then g + h update), each
//     wave runs two independent K-split accumulator chains to hide WMMA
//     RAW latency (scan is latency-bound: M=32 only)
//   - weights & activations bf16, state/accumulation fp32
// ---------------------------------------------------------------------------

constexpr int NB = 32;      // batch
constexpr int NS = 512;     // seq len
constexpr int NI = 256;     // input dim
constexpr int NH = 1024;    // hidden dim
constexpr int NO = 256;     // out dim
constexpr int NH3 = 3 * NH; // stacked gates

typedef __attribute__((ext_vector_type(16))) __bf16 v16bf;
typedef __attribute__((ext_vector_type(8)))  __bf16 v8bf;
typedef __attribute__((ext_vector_type(8)))  float  v8f;

__device__ __forceinline__ v8f wmma_bf16(v16bf a, v16bf b, v8f c) {
  // (neg_a, A, neg_b, B, c_mod, C, reuse_a, reuse_b)
  return __builtin_amdgcn_wmma_f32_16x16x32_bf16(false, a, false, b,
                                                 (short)0, c, false, false);
}

// A fragment: 16x32 (MxK) bf16 tile. ISA layout: lane%16 = M row;
// lanes 0-15 hold K slots {0..7, 16..23}, lanes 16-31 hold {8..15, 24..31}.
// K contiguous in memory -> two 16B loads per lane.
__device__ __forceinline__ v16bf load_frag_a(const __bf16* __restrict__ A,
                                             int ldk, int m_base, int k_base,
                                             int lane) {
  const int m  = m_base + (lane & 15);
  const int hi = lane >> 4;                 // 0 or 1
  const __bf16* row = A + (size_t)m * ldk + k_base + hi * 8;
  v8bf lo = *(const v8bf*)(row);            // K = k_base + hi*8 .. +7
  v8bf hv = *(const v8bf*)(row + 16);       // K = k_base + 16 + hi*8 .. +7
  v16bf r;
#pragma unroll
  for (int i = 0; i < 8; ++i) { r[i] = lo[i]; r[i + 8] = hv[i]; }
  return r;
}

// B fragment: 32x16 (KxN) bf16 tile, B[k][n] = W[n][k] with W row-major (N,K).
// lane%16 = N column (a row of W); lanes 0-15 cover K {0..7, 16..23} of the k-block,
// lanes 16-31 cover K {8..15, 24..31}. Two 16B loads per lane.
__device__ __forceinline__ v16bf load_frag_b(const __bf16* __restrict__ W,
                                             int ldk, int n_base, int k_base,
                                             int lane) {
  const int n  = n_base + (lane & 15);
  const int k0 = k_base + ((lane >> 4) << 3);
  const __bf16* row = W + (size_t)n * ldk + k0;
  v8bf lo = *(const v8bf*)(row);
  v8bf hv = *(const v8bf*)(row + 16);
  v16bf r;
#pragma unroll
  for (int i = 0; i < 8; ++i) { r[i] = lo[i]; r[i + 8] = hv[i]; }
  return r;
}

// C/D 16x16 f32: lane<16 -> n=lane, m=i ; lane>=16 -> n=lane-16, m=i+8.

// ---------------------------------------------------------------------------
// Generic GEMM:  out[m,n] = sum_k A[m,k] * W[n,k]  (+ bias[n])
// A: (M,K) bf16 row-major. W: (N,K) bf16 row-major.
// One 16(M) x 32(N) macro-tile per wave: A fragment reused for 2 WMMAs,
// two independent accumulator chains. N must be a multiple of 32.
// ---------------------------------------------------------------------------
__global__ void gemm_wmma(const __bf16* __restrict__ A,
                          const __bf16* __restrict__ W,
                          float* __restrict__ outF,
                          __bf16* __restrict__ outB,
                          const float* __restrict__ bias,
                          int M, int N, int K) {
  const int lane = threadIdx.x & 31;
  const int wave = threadIdx.x >> 5;
  const int tn2 = N >> 5;                    // pairs of N-tiles
  const int ntiles = (M >> 4) * tn2;
  const int tile = blockIdx.x * 8 + wave;
  if (tile >= ntiles) return;
  const int mt = tile / tn2;
  const int np = tile % tn2;
  const int n0 = np * 32;

  v8f acc0 = {}, acc1 = {};
  const int ksteps = K >> 5;
  for (int kt = 0; kt < ksteps; ++kt) {
    v16bf a  = load_frag_a(A, K, mt * 16, kt * 32, lane);
    v16bf b0 = load_frag_b(W, K, n0,      kt * 32, lane);
    v16bf b1 = load_frag_b(W, K, n0 + 16, kt * 32, lane);
    acc0 = wmma_bf16(a, b0, acc0);
    acc1 = wmma_bf16(a, b1, acc1);
  }

  const int nl = lane & 15;
  const int mb = (lane >> 4) * 8;
  const int na = n0 + nl;
  const int nbv = n0 + 16 + nl;
  const float bva = bias ? bias[na] : 0.0f;
  const float bvb = bias ? bias[nbv] : 0.0f;
#pragma unroll
  for (int i = 0; i < 8; ++i) {
    const int m = mt * 16 + mb + i;
    const float v0 = acc0[i] + bva;
    const float v1 = acc1[i] + bvb;
    if (outF) {
      outF[(size_t)m * N + na]  = v0;
      outF[(size_t)m * N + nbv] = v1;
    } else {
      outB[(size_t)m * N + na]  = (__bf16)v0;
      outB[(size_t)m * N + nbv] = (__bf16)v1;
    }
  }
}

// ---------------------------------------------------------------------------
// GRU scan phase 1: z = sigmoid(xz + h@Whz^T + bz), r = sigmoid(xr + h@Whr^T + br)
// Writes z (fp32) and rh = r*h (bf16, A-operand for phase 2).
// Whs: stacked (3H, H): rows [0,H)=Whz, [H,2H)=Whr, [2H,3H)=Whg.
// proj: (B*S, 3H) bf16 precomputed input projections (no bias).
// Grid: 32 blocks x 256 thr = 256 waves = 2 gates x 2 Mtiles x 64 Ntiles.
// K loop split into 2 independent accumulator chains (latency hiding).
// ---------------------------------------------------------------------------
__global__ void gru_phase1(const __bf16* __restrict__ hb,   // (B,H) bf16 h_{t-1}
                           const float* __restrict__ h,     // (B,H) f32  h_{t-1}
                           const __bf16* __restrict__ Whs,  // (3H,H)
                           const __bf16* __restrict__ proj, // (B*S,3H)
                           const float* __restrict__ bz,
                           const float* __restrict__ br,
                           float* __restrict__ z,           // (B,H)
                           __bf16* __restrict__ rh,         // (B,H)
                           int t) {
  const int lane = threadIdx.x & 31;
  const int wave = threadIdx.x >> 5;
  const int tile = blockIdx.x * 8 + wave;   // 0..255
  const int gate = tile >> 7;               // 0 = z, 1 = r (wave-uniform)
  const int rem  = tile & 127;
  const int mt = rem >> 6;                  // 0..1
  const int nt = rem & 63;                  // 0..63

  const __bf16* W = Whs + (size_t)gate * NH * NH;
  v8f acc0 = {}, acc1 = {};
#pragma unroll 4
  for (int kt = 0; kt < NH / 32; kt += 2) {
    v16bf a0 = load_frag_a(hb, NH, mt * 16, kt * 32, lane);
    v16bf b0 = load_frag_b(W, NH, nt * 16, kt * 32, lane);
    v16bf a1 = load_frag_a(hb, NH, mt * 16, (kt + 1) * 32, lane);
    v16bf b1 = load_frag_b(W, NH, nt * 16, (kt + 1) * 32, lane);
    acc0 = wmma_bf16(a0, b0, acc0);
    acc1 = wmma_bf16(a1, b1, acc1);
  }

  const int j  = nt * 16 + (lane & 15);
  const int mb = (lane >> 4) * 8;
  const float bias = gate ? br[j] : bz[j];
#pragma unroll
  for (int i = 0; i < 8; ++i) {
    const int b = mt * 16 + mb + i;
    const float xin = (float)proj[(size_t)(b * NS + t) * NH3 + gate * NH + j];
    const float pre = acc0[i] + acc1[i] + xin + bias;
    const float v = 1.0f / (1.0f + __expf(-pre));
    if (gate == 0) z[b * NH + j] = v;
    else           rh[b * NH + j] = (__bf16)(v * h[b * NH + j]);
  }
}

// ---------------------------------------------------------------------------
// GRU scan phase 2: g = tanh(xg + (r*h)@Whg^T + bg); h' = z*h + (1-z)*g.
// Writes h (fp32, in place), hb (bf16 for next step's A), y[:,t,:] (bf16).
// Grid: 16 blocks x 256 thr = 128 waves = 2 Mtiles x 64 Ntiles.
// ---------------------------------------------------------------------------
__global__ void gru_phase2(const __bf16* __restrict__ rh,   // (B,H)
                           const __bf16* __restrict__ Whs,  // (3H,H)
                           const __bf16* __restrict__ proj, // (B*S,3H)
                           const float* __restrict__ bg,
                           const float* __restrict__ z,     // (B,H)
                           float* __restrict__ h,           // (B,H) in/out
                           __bf16* __restrict__ hb,         // (B,H) out
                           __bf16* __restrict__ y,          // (B,S,H) out
                           int t) {
  const int lane = threadIdx.x & 31;
  const int wave = threadIdx.x >> 5;
  const int tile = blockIdx.x * 8 + wave;   // 0..127
  const int mt = tile >> 6;
  const int nt = tile & 63;

  const __bf16* W = Whs + (size_t)2 * NH * NH;   // Whg
  v8f acc0 = {}, acc1 = {};
#pragma unroll 4
  for (int kt = 0; kt < NH / 32; kt += 2) {
    v16bf a0 = load_frag_a(rh, NH, mt * 16, kt * 32, lane);
    v16bf b0 = load_frag_b(W, NH, nt * 16, kt * 32, lane);
    v16bf a1 = load_frag_a(rh, NH, mt * 16, (kt + 1) * 32, lane);
    v16bf b1 = load_frag_b(W, NH, nt * 16, (kt + 1) * 32, lane);
    acc0 = wmma_bf16(a0, b0, acc0);
    acc1 = wmma_bf16(a1, b1, acc1);
  }

  const int j  = nt * 16 + (lane & 15);
  const int mb = (lane >> 4) * 8;
  const float bias = bg[j];
#pragma unroll
  for (int i = 0; i < 8; ++i) {
    const int b = mt * 16 + mb + i;
    const float xin = (float)proj[(size_t)(b * NS + t) * NH3 + 2 * NH + j];
    const float g = tanhf(acc0[i] + acc1[i] + xin + bias);
    const float zz = z[b * NH + j];
    const float hn = zz * h[b * NH + j] + (1.0f - zz) * g;
    h[b * NH + j]  = hn;
    hb[b * NH + j] = (__bf16)hn;
    y[(size_t)(b * NS + t) * NH + j] = (__bf16)hn;
  }
}

// ---------------------------------------------------------------------------
// Small utility kernels
// ---------------------------------------------------------------------------
__global__ void cvt_f32_bf16(const float* __restrict__ src,
                             __bf16* __restrict__ dst, int n) {
  int i = blockIdx.x * blockDim.x + threadIdx.x;
  const int stride = gridDim.x * blockDim.x;
  for (; i < n; i += stride) dst[i] = (__bf16)src[i];
}

__global__ void zero_state(float* __restrict__ h, __bf16* __restrict__ hb, int n) {
  int i = blockIdx.x * blockDim.x + threadIdx.x;
  const int stride = gridDim.x * blockDim.x;
  for (; i < n; i += stride) { h[i] = 0.0f; hb[i] = (__bf16)0.0f; }
}

// hidden[b, l, j] <- h[b, j]  (hidden is (B, 2, H) fp32)
__global__ void copy_hidden(const float* __restrict__ h,
                            float* __restrict__ hidden, int l) {
  int i = blockIdx.x * blockDim.x + threadIdx.x;
  const int stride = gridDim.x * blockDim.x;
  for (; i < NB * NH; i += stride) {
    const int b = i / NH, j = i % NH;
    hidden[((size_t)b * 2 + l) * NH + j] = h[i];
  }
}

// ---------------------------------------------------------------------------
extern "C" void kernel_launch(void* const* d_in, const int* in_sizes, int n_in,
                              void* d_out, int out_size, void* d_ws, size_t ws_size,
                              hipStream_t stream) {
  (void)in_sizes; (void)n_in; (void)out_size; (void)ws_size;

  const float* x     = (const float*)d_in[0];
  const float* Wxz0  = (const float*)d_in[1];
  const float* bxz0  = (const float*)d_in[2];
  const float* Whz0  = (const float*)d_in[3];
  const float* Wxr0  = (const float*)d_in[4];
  const float* bxr0  = (const float*)d_in[5];
  const float* Whr0  = (const float*)d_in[6];
  const float* Wxg0  = (const float*)d_in[7];
  const float* bxg0  = (const float*)d_in[8];
  const float* Whg0  = (const float*)d_in[9];
  const float* Wxz1  = (const float*)d_in[10];
  const float* bxz1  = (const float*)d_in[11];
  const float* Whz1  = (const float*)d_in[12];
  const float* Wxr1  = (const float*)d_in[13];
  const float* bxr1  = (const float*)d_in[14];
  const float* Whr1  = (const float*)d_in[15];
  const float* Wxg1  = (const float*)d_in[16];
  const float* bxg1  = (const float*)d_in[17];
  const float* Whg1  = (const float*)d_in[18];
  const float* Wout  = (const float*)d_in[19];
  const float* bout  = (const float*)d_in[20];

  float* out    = (float*)d_out;                        // (B, S, O)
  float* hidden = (float*)d_out + (size_t)NB * NS * NO; // (B, 2, H)

  // ---- carve workspace (256B aligned slabs) ----
  char* p = (char*)d_ws;
  auto alloc = [&p](size_t bytes) -> void* {
    void* r = (void*)p;
    p += (bytes + 255) & ~(size_t)255;
    return r;
  };
  __bf16* xb    = (__bf16*)alloc((size_t)NB * NS * NI * 2);  // x in bf16
  __bf16* Wx0s  = (__bf16*)alloc((size_t)NH3 * NI * 2);      // [Wxz0;Wxr0;Wxg0]
  __bf16* Wh0s  = (__bf16*)alloc((size_t)NH3 * NH * 2);      // [Whz0;Whr0;Whg0]
  __bf16* Wx1s  = (__bf16*)alloc((size_t)NH3 * NH * 2);
  __bf16* Wh1s  = (__bf16*)alloc((size_t)NH3 * NH * 2);
  __bf16* WoutB = (__bf16*)alloc((size_t)NO * NH * 2);
  __bf16* proj  = (__bf16*)alloc((size_t)NB * NS * NH3 * 2); // reused for L0/L1
  __bf16* y0    = (__bf16*)alloc((size_t)NB * NS * NH * 2);
  __bf16* y1    = (__bf16*)alloc((size_t)NB * NS * NH * 2);
  float*  h     = (float*)alloc((size_t)NB * NH * 4);
  __bf16* hb    = (__bf16*)alloc((size_t)NB * NH * 2);
  __bf16* rh    = (__bf16*)alloc((size_t)NB * NH * 2);
  float*  zbuf  = (float*)alloc((size_t)NB * NH * 4);

  const int HI = NH * NI;   // 262144
  const int HH = NH * NH;   // 1048576

  // ---- convert inputs / stack weights to bf16 ----
  cvt_f32_bf16<<<2048, 256, 0, stream>>>(x, xb, NB * NS * NI);
  cvt_f32_bf16<<<512, 256, 0, stream>>>(Wxz0, Wx0s + 0 * (size_t)HI, HI);
  cvt_f32_bf16<<<512, 256, 0, stream>>>(Wxr0, Wx0s + 1 * (size_t)HI, HI);
  cvt_f32_bf16<<<512, 256, 0, stream>>>(Wxg0, Wx0s + 2 * (size_t)HI, HI);
  cvt_f32_bf16<<<1024, 256, 0, stream>>>(Whz0, Wh0s + 0 * (size_t)HH, HH);
  cvt_f32_bf16<<<1024, 256, 0, stream>>>(Whr0, Wh0s + 1 * (size_t)HH, HH);
  cvt_f32_bf16<<<1024, 256, 0, stream>>>(Whg0, Wh0s + 2 * (size_t)HH, HH);
  cvt_f32_bf16<<<1024, 256, 0, stream>>>(Wxz1, Wx1s + 0 * (size_t)HH, HH);
  cvt_f32_bf16<<<1024, 256, 0, stream>>>(Wxr1, Wx1s + 1 * (size_t)HH, HH);
  cvt_f32_bf16<<<1024, 256, 0, stream>>>(Wxg1, Wx1s + 2 * (size_t)HH, HH);
  cvt_f32_bf16<<<1024, 256, 0, stream>>>(Whz1, Wh1s + 0 * (size_t)HH, HH);
  cvt_f32_bf16<<<1024, 256, 0, stream>>>(Whr1, Wh1s + 1 * (size_t)HH, HH);
  cvt_f32_bf16<<<1024, 256, 0, stream>>>(Whg1, Wh1s + 2 * (size_t)HH, HH);
  cvt_f32_bf16<<<512, 256, 0, stream>>>(Wout, WoutB, NO * NH);

  const int M = NB * NS;                            // 16384
  const int tilesProj = (M / 16) * (NH3 / 32);      // 1024 * 96 macro-tiles

  // ---- layer 0 ----
  gemm_wmma<<<tilesProj / 8, 256, 0, stream>>>(xb, Wx0s, nullptr, proj,
                                               nullptr, M, NH3, NI);
  zero_state<<<64, 256, 0, stream>>>(h, hb, NB * NH);
  for (int t = 0; t < NS; ++t) {
    gru_phase1<<<32, 256, 0, stream>>>(hb, h, Wh0s, proj, bxz0, bxr0, zbuf, rh, t);
    gru_phase2<<<16, 256, 0, stream>>>(rh, Wh0s, proj, bxg0, zbuf, h, hb, y0, t);
  }
  copy_hidden<<<64, 256, 0, stream>>>(h, hidden, 0);

  // ---- layer 1 ----
  gemm_wmma<<<tilesProj / 8, 256, 0, stream>>>(y0, Wx1s, nullptr, proj,
                                               nullptr, M, NH3, NH);
  zero_state<<<64, 256, 0, stream>>>(h, hb, NB * NH);
  for (int t = 0; t < NS; ++t) {
    gru_phase1<<<32, 256, 0, stream>>>(hb, h, Wh1s, proj, bxz1, bxr1, zbuf, rh, t);
    gru_phase2<<<16, 256, 0, stream>>>(rh, Wh1s, proj, bxg1, zbuf, h, hb, y1, t);
  }
  copy_hidden<<<64, 256, 0, stream>>>(h, hidden, 1);

  // ---- output projection: out = y1 @ Wout^T + bout ----
  const int tilesOut = (M / 16) * (NO / 32);        // 1024 * 8 macro-tiles
  gemm_wmma<<<tilesOut / 8, 256, 0, stream>>>(y1, WoutB, out, nullptr,
                                              bout, M, NO, NH);
}
